// SelfAttention_5652176961757
// MI455X (gfx1250) — matrix operand order repaired and checked by reading, not compiled, not run on hardware
//
#include <hip/hip_runtime.h>

// ---------------------------------------------------------------------------
// Problem sizes (fixed by the reference)
// ---------------------------------------------------------------------------
static constexpr int BB = 2;        // batch
static constexpr int LL = 2048;     // sequence
static constexpr int CC = 1024;     // channels
static constexpr int HH = 16;       // heads
static constexpr int CH = 64;       // head dim
static constexpr int M1 = BB * LL;  // 4096 tokens

#define DEV __device__ __forceinline__

typedef __bf16 bf16;
typedef bf16  bf16x2  __attribute__((ext_vector_type(2)));
typedef bf16  bf16x4  __attribute__((ext_vector_type(4)));
typedef bf16  bf16x8  __attribute__((ext_vector_type(8)));
typedef bf16  bf16x16 __attribute__((ext_vector_type(16)));
typedef float f32x8   __attribute__((ext_vector_type(8)));

#if defined(__gfx1250__)
typedef __attribute__((address_space(3))) char as3char;
#endif

// ---------------------------------------------------------------------------
// bf16 <-> f32 (bit-level, RNE) — avoids any dependency on __bf16 conversions
// ---------------------------------------------------------------------------
DEV bf16 f2bf(float f) {
    unsigned u = __builtin_bit_cast(unsigned, f);
    u += 0x7FFFu + ((u >> 16) & 1u);
    unsigned short h = (unsigned short)(u >> 16);
    return __builtin_bit_cast(bf16, h);
}
DEV float bf2f(bf16 b) {
    unsigned short h = __builtin_bit_cast(unsigned short, b);
    unsigned u = ((unsigned)h) << 16;
    return __builtin_bit_cast(float, u);
}

// ---------------------------------------------------------------------------
// WMMA: D = A(16x32 bf16) x B(32x16 bf16) + C(16x16 f32)
// ---------------------------------------------------------------------------
DEV f32x8 wmma_bf16(bf16x16 a, bf16x16 b, f32x8 c) {
    return __builtin_amdgcn_wmma_f32_16x16x32_bf16(
        /*neg_a=*/false, a, /*neg_b=*/false, b,
        /*c_mod=*/(short)0, c, /*reuse_a=*/false, /*reuse_b=*/false);
}

// A-fragment (16x32, MxK) from a row-major [rows][stride] bf16 matrix.
// lane: row = lane&15; K chunks {kb..kb+7} and {kb+16..kb+23}, kb = lane<16?0:8
DEV bf16x16 frag_a(const bf16* base, int stride, int lane) {
    const bf16* p = base + (size_t)(lane & 15) * stride + ((lane & 16) ? 8 : 0);
    bf16x8 lo = *(const bf16x8*)p;
    bf16x8 hi = *(const bf16x8*)(p + 16);
    return __builtin_shufflevector(lo, hi, 0, 1, 2, 3, 4, 5, 6, 7,
                                   8, 9, 10, 11, 12, 13, 14, 15);
}

// B-fragment (32x16, KxN) where column n of B is row n of a row-major [N][stride]
// matrix. lane: n = lane&15; K chunks {kb..kb+7} and {kb+16..kb+23}, kb = lane<16?0:8
DEV bf16x16 frag_b(const bf16* base, int stride, int lane) {
    const bf16* p = base + (size_t)(lane & 15) * stride + ((lane & 16) ? 8 : 0);
    bf16x8 lo = *(const bf16x8*)p;
    bf16x8 hi = *(const bf16x8*)(p + 16);
    return __builtin_shufflevector(lo, hi, 0, 1, 2, 3, 4, 5, 6, 7,
                                   8, 9, 10, 11, 12, 13, 14, 15);
}

// Async global->LDS staging of one 16B chunk (gfx1250 ASYNCcnt path).
DEV void stage16(const bf16* g, bf16* l) {
#if defined(__gfx1250__)
    unsigned lofs = (unsigned)(__UINTPTR_TYPE__)(as3char*)l;
    unsigned long long ga = (unsigned long long)(__UINTPTR_TYPE__)g;
    asm volatile("global_load_async_to_lds_b128 %0, %1, off"
                 :
                 : "v"(lofs), "v"(ga)
                 : "memory");
#else
    *(bf16x8*)l = *(const bf16x8*)g;
#endif
}
DEV void stage_wait() {
#if defined(__gfx1250__)
    asm volatile("s_wait_asynccnt 0x0" ::: "memory");
#endif
}

// Reductions across 16-lane halves (matches C-fragment row layout)
DEV float redmax16(float v) {
#pragma unroll
    for (int m = 1; m <= 8; m <<= 1) v = fmaxf(v, __shfl_xor(v, m));
    return v;
}
DEV float redsum16(float v) {
#pragma unroll
    for (int m = 1; m <= 8; m <<= 1) v += __shfl_xor(v, m);
    return v;
}

// ---------------------------------------------------------------------------
// K0: f32 -> bf16 bulk convert (4 elems/thread)
// ---------------------------------------------------------------------------
__global__ __launch_bounds__(256) void cvt_f32_bf16(const float* __restrict__ in,
                                                    bf16* __restrict__ out, int n4) {
    int i = blockIdx.x * 256 + threadIdx.x;
    if (i >= n4) return;
    float4 v = ((const float4*)in)[i];
    bf16x4 o;
    o[0] = f2bf(v.x); o[1] = f2bf(v.y); o[2] = f2bf(v.z); o[3] = f2bf(v.w);
    ((bf16x4*)out)[i] = o;
}

// K0b: assemble concat(q_bias, 0, v_bias) -> float[3072]
__global__ __launch_bounds__(256) void build_bias(const float* __restrict__ qb,
                                                  const float* __restrict__ vb,
                                                  float* __restrict__ biasf) {
    int n = blockIdx.x * 256 + threadIdx.x;
    if (n >= 3 * CC) return;
    int which = n >> 10, idx = n & (CC - 1);
    float v = 0.f;
    if (which == 0) v = qb[idx];
    else if (which == 2) v = vb[idx];
    biasf[n] = v;
}

// ---------------------------------------------------------------------------
// GEMM: out[M][N] = A[M][K](bf16) * W[N][K]^T(bf16) + bias[N]
// 128x64 tile per block, 8 waves (each 16 rows x 64 cols), K-step 32.
// Tiles staged global->LDS with async-to-LDS (ASYNCcnt) on gfx1250.
// ---------------------------------------------------------------------------
template <bool OUT_BF16>
__global__ __launch_bounds__(256) void gemm_bf16_kernel(
    const bf16* __restrict__ A, const bf16* __restrict__ W,
    const float* __restrict__ bias, float* __restrict__ outF,
    bf16* __restrict__ outB, int M, int N, int K) {
    __shared__ __attribute__((aligned(16))) bf16 As[128 * 32];
    __shared__ __attribute__((aligned(16))) bf16 Ws[64 * 32];
    const int tid  = threadIdx.x;
    const int lane = tid & 31;
    const int wave = tid >> 5;
    const int nbn  = N >> 6;
    const int bm   = (int)(blockIdx.x / nbn) << 7;
    const int bn   = (int)(blockIdx.x % nbn) << 6;
    const int arow = tid >> 1;        // 0..127
    const int acol = (tid & 1) << 4;  // 0 or 16
    const int wrow = tid >> 2;        // 0..63
    const int wcol = (tid & 3) << 3;  // 0,8,16,24

    const bf16* Ag = A + (size_t)(bm + arow) * K + acol;
    const bf16* Wg = W + (size_t)(bn + wrow) * K + wcol;
    bf16* Al = As + arow * 32 + acol;
    bf16* Wl = Ws + wrow * 32 + wcol;

    f32x8 acc[4];
#pragma unroll
    for (int j = 0; j < 4; ++j) acc[j] = (f32x8){0, 0, 0, 0, 0, 0, 0, 0};

    for (int k0 = 0; k0 < K; k0 += 32) {
        __syncthreads();  // previous tile fully consumed
        stage16(Ag + k0, Al);
        stage16(Ag + k0 + 8, Al + 8);
        stage16(Wg + k0, Wl);
        stage_wait();
        __syncthreads();  // tile visible to all waves
        bf16x16 af = frag_a(As + (wave << 4) * 32, 32, lane);
#pragma unroll
        for (int j = 0; j < 4; ++j)
            acc[j] = wmma_bf16(af, frag_b(Ws + (j << 4) * 32, 32, lane), acc[j]);
    }

    const int mofs = (lane & 16) ? 8 : 0;
    const int nlo  = lane & 15;
#pragma unroll
    for (int j = 0; j < 4; ++j) {
        int ncol = bn + j * 16 + nlo;
        float bv = bias[ncol];
#pragma unroll
        for (int r = 0; r < 8; ++r) {
            size_t idx = (size_t)(bm + (wave << 4) + r + mofs) * N + ncol;
            float v = acc[j][r] + bv;
            if (OUT_BF16) outB[idx] = f2bf(v);
            else          outF[idx] = v;
        }
    }
}

// ---------------------------------------------------------------------------
// K2: per-(b,h,l) row: L2-norm q (x exp-scale), L2-norm k, scatter v^T.
// One wave per row of 64 elements (2 per lane).
// Layouts: Q,K -> [head][L][64] bf16 ; Vt -> [head][64][L] bf16
// ---------------------------------------------------------------------------
__global__ __launch_bounds__(256) void norm_split_kernel(
    const bf16* __restrict__ qkv, const float* __restrict__ logit,
    bf16* __restrict__ Q, bf16* __restrict__ Kx, bf16* __restrict__ Vt) {
    const int lane = threadIdx.x & 31;
    const int wave = threadIdx.x >> 5;
    const int row  = blockIdx.x * 8 + wave;  // 0 .. B*H*L-1
    const int l    = row & (LL - 1);
    const int head = row >> 11;              // row / L
    const int h    = head & (HH - 1);
    const int b    = head >> 4;

    const bf16* base = qkv + ((size_t)b * LL + l) * (3 * CC) + h * CH + 2 * lane;
    bf16x2 qv = *(const bf16x2*)(base);
    bf16x2 kv = *(const bf16x2*)(base + CC);
    bf16x2 vv = *(const bf16x2*)(base + 2 * CC);
    float q0 = bf2f(qv[0]), q1 = bf2f(qv[1]);
    float k0 = bf2f(kv[0]), k1 = bf2f(kv[1]);
    float sq = q0 * q0 + q1 * q1;
    float sk = k0 * k0 + k1 * k1;
#pragma unroll
    for (int m = 16; m; m >>= 1) {
        sq += __shfl_xor(sq, m);
        sk += __shfl_xor(sk, m);
    }
    float sm = __expf(fminf(logit[h], 4.6051702f));  // exp(min(., log 100))
    float iq = sm / fmaxf(sqrtf(sq), 1e-12f);
    float ik = 1.f / fmaxf(sqrtf(sk), 1e-12f);

    size_t o = ((size_t)head * LL + l) * CH + 2 * lane;
    bf16x2 qo; qo[0] = f2bf(q0 * iq); qo[1] = f2bf(q1 * iq);
    bf16x2 ko; ko[0] = f2bf(k0 * ik); ko[1] = f2bf(k1 * ik);
    *(bf16x2*)(Q + o)  = qo;
    *(bf16x2*)(Kx + o) = ko;

    size_t vo = ((size_t)head * CH + 2 * lane) * LL + l;  // transposed
    Vt[vo]      = vv[0];
    Vt[vo + LL] = vv[1];
}

// ---------------------------------------------------------------------------
// K3: fused flash attention. 1 wave = one 16-query tile; 32 keys / iteration.
// Per iter: 4 WMMA (Q.K^T, c=64 as 2 chained K=32 x 2 key tiles) + softmax
// update + P relayout via per-wave LDS slab + 4 WMMA (P.V over 64-wide V).
// Writes O (bf16) in [B,L,H*c] layout ready for the projection GEMM.
// ---------------------------------------------------------------------------
__global__ __launch_bounds__(128) void attn_kernel(
    const bf16* __restrict__ Q, const bf16* __restrict__ Kx,
    const bf16* __restrict__ Vt, const float* __restrict__ bias,
    bf16* __restrict__ O) {
    __shared__ __attribute__((aligned(16))) bf16 Pl[4][16 * 32];
    const int lane = threadIdx.x & 31;
    const int wave = threadIdx.x >> 5;
    const int qt   = blockIdx.x * 4 + wave;   // global 16-query tile id
    const int head = qt >> 7;                 // 128 tiles per head
    const int q0   = (qt & 127) << 4;
    const int h    = head & (HH - 1);
    const int b    = head >> 4;

    const bf16* Qp = Q  + ((size_t)head * LL + q0) * CH;
    const bf16* Kp = Kx + (size_t)head * LL * CH;
    const bf16* Vp = Vt + (size_t)head * CH * LL;

    const int mofs = (lane & 16) ? 8 : 0;  // C-frag row offset for this lane half
    const int nlo  = lane & 15;

    bf16x16 qa0 = frag_a(Qp, CH, lane);       // c 0..31
    bf16x16 qa1 = frag_a(Qp + 32, CH, lane);  // c 32..63

    f32x8 acc0 = {0, 0, 0, 0, 0, 0, 0, 0};
    f32x8 acc1 = acc0, acc2 = acc0, acc3 = acc0;
    float mrow[8], lrow[8];
#pragma unroll
    for (int r = 0; r < 8; ++r) { mrow[r] = -3.0e30f; lrow[r] = 0.f; }

    for (int k0 = 0; k0 < LL; k0 += 32) {
        // hint the next K/V tiles into cache (global_prefetch_b8)
        int kn = (k0 + 32) & (LL - 1);
        __builtin_prefetch(Kp + (size_t)kn * CH, 0, 1);
        __builtin_prefetch(Vp + kn, 0, 1);

        const f32x8 z = {0, 0, 0, 0, 0, 0, 0, 0};
        // scores for key tiles [k0,k0+16) and [k0+16,k0+32)
        f32x8 s0 = wmma_bf16(qa0, frag_b(Kp + (size_t)k0 * CH, CH, lane), z);
        s0 = wmma_bf16(qa1, frag_b(Kp + (size_t)k0 * CH + 32, CH, lane), s0);
        f32x8 s1 = wmma_bf16(qa0, frag_b(Kp + (size_t)(k0 + 16) * CH, CH, lane), z);
        s1 = wmma_bf16(qa1, frag_b(Kp + (size_t)(k0 + 16) * CH + 32, CH, lane), s1);

        // + attn_bias[q, k]
        const float* bp = bias + (size_t)(q0 + mofs) * LL + k0 + nlo;
#pragma unroll
        for (int r = 0; r < 8; ++r) {
            s0[r] += bp[(size_t)r * LL];
            s1[r] += bp[(size_t)r * LL + 16];
        }

        // online softmax
#pragma unroll
        for (int r = 0; r < 8; ++r) {
            float mx = redmax16(fmaxf(s0[r], s1[r]));
            float mn = fmaxf(mrow[r], mx);
            float alpha = __expf(mrow[r] - mn);
            mrow[r] = mn;
            float p0 = __expf(s0[r] - mn);
            float p1 = __expf(s1[r] - mn);
            s0[r] = p0; s1[r] = p1;
            lrow[r] = lrow[r] * alpha + redsum16(p0 + p1);
            acc0[r] *= alpha; acc1[r] *= alpha;
            acc2[r] *= alpha; acc3[r] *= alpha;
        }

        // relayout P (C-layout -> A-layout) through the wave's LDS slab
        bf16* ps = &Pl[wave][0];
#pragma unroll
        for (int r = 0; r < 8; ++r) {
            int mm = r + mofs;
            ps[mm * 32 + nlo]      = f2bf(s0[r]);
            ps[mm * 32 + 16 + nlo] = f2bf(s1[r]);
        }
        bf16x16 pa = frag_a(ps, 32, lane);  // same-wave DS ops are in order

        // acc += P(16x32) . V(32x64), V^T rows are contiguous in keys
        acc0 = wmma_bf16(pa, frag_b(Vp + k0, LL, lane), acc0);
        acc1 = wmma_bf16(pa, frag_b(Vp + (size_t)16 * LL + k0, LL, lane), acc1);
        acc2 = wmma_bf16(pa, frag_b(Vp + (size_t)32 * LL + k0, LL, lane), acc2);
        acc3 = wmma_bf16(pa, frag_b(Vp + (size_t)48 * LL + k0, LL, lane), acc3);
    }

    // epilogue: O[b, q, h*64 + c] = acc / l
    bf16* op = O + ((size_t)b * LL + q0) * CC + h * CH;
#pragma unroll
    for (int r = 0; r < 8; ++r) {
        float inv = 1.f / lrow[r];
        size_t ro = (size_t)(r + mofs) * CC;
        op[ro + 0 * 16 + nlo] = f2bf(acc0[r] * inv);
        op[ro + 1 * 16 + nlo] = f2bf(acc1[r] * inv);
        op[ro + 2 * 16 + nlo] = f2bf(acc2[r] * inv);
        op[ro + 3 * 16 + nlo] = f2bf(acc3[r] * inv);
    }
}

// ---------------------------------------------------------------------------
// Host side
// ---------------------------------------------------------------------------
extern "C" void kernel_launch(void* const* d_in, const int* in_sizes, int n_in,
                              void* d_out, int out_size, void* d_ws, size_t ws_size,
                              hipStream_t stream) {
    (void)in_sizes; (void)n_in; (void)out_size; (void)ws_size;
    const float* x      = (const float*)d_in[0];
    const float* abias  = (const float*)d_in[1];
    const float* w_qkv  = (const float*)d_in[2];
    const float* q_bias = (const float*)d_in[3];
    const float* v_bias = (const float*)d_in[4];
    const float* logit  = (const float*)d_in[5];
    const float* w_proj = (const float*)d_in[6];
    const float* b_proj = (const float*)d_in[7];
    float* out = (float*)d_out;

    // workspace carve-out (256B aligned); total ~76 MB
    char* ws = (char*)d_ws;
    size_t off = 0;
    auto take = [&](size_t bytes) {
        char* p = ws + off;
        off += (bytes + 255) & ~(size_t)255;
        return p;
    };
    bf16*  xb     = (bf16*)take((size_t)M1 * CC * 2);          // x in bf16
    bf16*  wqkvb  = (bf16*)take((size_t)3 * CC * CC * 2);      // w_qkv bf16
    bf16*  wprojb = (bf16*)take((size_t)CC * CC * 2);          // w_proj bf16
    float* biasf  = (float*)take((size_t)3 * CC * 4);          // qkv bias
    bf16*  qkvb   = (bf16*)take((size_t)M1 * 3 * CC * 2);      // raw qkv bf16
    bf16*  Qb     = (bf16*)take((size_t)BB * HH * LL * CH * 2);
    bf16*  Kb     = (bf16*)take((size_t)BB * HH * LL * CH * 2);
    bf16*  Vtb    = (bf16*)take((size_t)BB * HH * CH * LL * 2);
    bf16*  Ob     = (bf16*)take((size_t)M1 * CC * 2);

    // K0: conversions + bias assembly
    cvt_f32_bf16<<<(M1 * CC / 4 + 255) / 256, 256, 0, stream>>>(x, xb, M1 * CC / 4);
    cvt_f32_bf16<<<(3 * CC * CC / 4 + 255) / 256, 256, 0, stream>>>(w_qkv, wqkvb,
                                                                    3 * CC * CC / 4);
    cvt_f32_bf16<<<(CC * CC / 4 + 255) / 256, 256, 0, stream>>>(w_proj, wprojb,
                                                                CC * CC / 4);
    build_bias<<<(3 * CC + 255) / 256, 256, 0, stream>>>(q_bias, v_bias, biasf);

    // K1: qkv = x @ w_qkv^T + bias  (4096 x 3072 x 1024), 128x64 tiles
    gemm_bf16_kernel<true><<<(M1 / 128) * (3 * CC / 64), 256, 0, stream>>>(
        xb, wqkvb, biasf, nullptr, qkvb, M1, 3 * CC, CC);

    // K2: l2norm + scale, split into Q,K and transposed V
    norm_split_kernel<<<BB * HH * LL / 8, 256, 0, stream>>>(qkvb, logit, Qb, Kb, Vtb);

    // K3: fused flash attention (1024 blocks x 4 waves = 4096 query tiles)
    attn_kernel<<<BB * HH * (LL / 16) / 4, 128, 0, stream>>>(Qb, Kb, Vtb, abias, Ob);

    // K4: out = O @ w_proj^T + b_proj  (4096 x 1024 x 1024), 128x64 tiles
    gemm_bf16_kernel<false><<<(M1 / 128) * (CC / 64), 256, 0, stream>>>(
        Ob, wprojb, b_proj, out, nullptr, M1, CC, CC);
}
